// Seq2Seq_28071906247220
// MI455X (gfx1250) — matrix operand order repaired and checked by reading, not compiled, not run on hardware
//
#include <hip/hip_runtime.h>
#include <hip/hip_bf16.h>

#define HS 1536
#define VS 128
#define TPB 256
#define NWG 96

typedef __attribute__((ext_vector_type(16))) __bf16 v16bf;
typedef __attribute__((ext_vector_type(8)))  float  v8f;

struct U128 { unsigned x, y, z, w; };
union AU { v16bf v; U128 q[2]; };

struct Params {
    const int* tokens; int L; int ML;
    const float *emb_enc, *enc_bih, *enc_bhh;
    const float *emb_dec, *attn_b, *comb_b, *dec_bih, *dec_bhh, *out_b;
    const __bf16 *w_encWih, *w_encWhh, *w_decWih, *w_decWhh, *w_comb, *w_attn, *w_out;
    float *h0, *h1, *enc_outs, *logitsA, *ctx, *xvec, *logitsO;
    float *out_logp, *out_tok;
    unsigned* bar;
};

// ---------------------------------------------------------------------------
// One wave computes y[m0..m0+15] (over K range [k0,k0+kn)) of
// W(bf16,row-major) @ x(f32, staged in LDS) using V_WMMA_F32_16X16X32_BF16,
// with x placed in column N=0 of B.  B's zero lanes are hoisted out of the
// K loop (loop-invariant); lanes 0/16 rewrite their x halves each iteration.
// Results for the 16 rows are written to out16[0..15] by lanes 0 and 16.
// ---------------------------------------------------------------------------
__device__ inline void wmma_matvec_tile(const __bf16* __restrict__ W, int ldw,
                                        int m0, const float* __restrict__ xv,
                                        int k0, int kn, float* out16)
{
    const int lane = threadIdx.x & 31;
    const int half = lane >> 4;                       // 0: lanes 0-15, 1: lanes 16-31
    // A layout (16-bit A 16x32): lanes 0-15 hold M=0..15 K{0..7,16..23};
    // lanes 16-31 hold M=0..15 K{8..15,24..31}; 2 bf16 per VGPR.
    const __bf16* wrow = W + (size_t)(m0 + (lane & 15)) * (size_t)ldw + (half ? 8 : 0);
    const bool bfill = ((lane & 15) == 0);            // lanes 0 and 16 carry column N=0
    const float* xs0 = xv + (half << 3);              // lane0: x[k+0..7,k+16..23], lane16: x[k+8..15,k+24..31]
    v8f   c = {};
    v16bf b = {};                                     // zero columns 1..15 stay zero
#pragma unroll 4
    for (int k = k0; k < k0 + kn; k += 32) {
        AU a;
        a.q[0] = *reinterpret_cast<const U128*>(wrow + k);        // K +0..7  (or +8..15)
        a.q[1] = *reinterpret_cast<const U128*>(wrow + k + 16);   // K +16..23(or +24..31)
        if (bfill) {
            const float* xs = xs0 + k;                // LDS reads (ds path)
#pragma unroll
            for (int e = 0; e < 16; ++e) b[e] = (__bf16)xs[e + (e & 8)];
        }
        c = __builtin_amdgcn_wmma_f32_16x16x32_bf16(false, a.v, false, b,
                                                    (short)0, c, false, false);
    }
    // D layout: column N=0 lives in lane0 (M=0..7 in v0..7) and lane16 (M=8..15)
    if (bfill) {
#pragma unroll
        for (int r = 0; r < 8; ++r) out16[(half << 3) + r] = c[r];
    }
}

// ---------------------------------------------------------------------------
// Coalesced global->LDS staging of a vector by the whole workgroup.
// ---------------------------------------------------------------------------
__device__ inline void stage(float* __restrict__ dst, const float* __restrict__ src, int n)
{
    for (int i = threadIdx.x; i < n; i += TPB) dst[i] = src[i];
}

// ---------------------------------------------------------------------------
// Device-wide barrier (persistent kernel, NWG co-resident workgroups).
// ---------------------------------------------------------------------------
__device__ inline void gsync(unsigned* bar)
{
    __threadfence();          // release this thread's global writes
    __syncthreads();
    if (threadIdx.x == 0) {
        unsigned* cnt = bar;
        unsigned* gen = bar + 1;
        volatile unsigned* vgen = (volatile unsigned*)gen;
        unsigned g = *vgen;
        if (atomicAdd(cnt, 1u) == NWG - 1u) {
            atomicExch(cnt, 0u);
            __threadfence();
            atomicAdd(gen, 1u);
        } else {
            while (*vgen == g) __builtin_amdgcn_s_sleep(2);
        }
    }
    __syncthreads();
    __threadfence();          // acquire: invalidate before reading others' data
}

// ---------------------------------------------------------------------------
// Helper kernels: fp32 -> bf16 weight conversion and state init
// ---------------------------------------------------------------------------
__global__ void k_cvt(const float* __restrict__ s, __bf16* __restrict__ d, long long n)
{
    long long st = (long long)gridDim.x * blockDim.x;
    for (long long i = (long long)blockIdx.x * blockDim.x + threadIdx.x; i < n; i += st)
        d[i] = (__bf16)s[i];
}

__global__ void k_cvt_pad(const float* __restrict__ s, __bf16* __restrict__ d,
                          int srows, int drows, int cols)
{
    long long n = (long long)drows * cols;
    long long st = (long long)gridDim.x * blockDim.x;
    for (long long i = (long long)blockIdx.x * blockDim.x + threadIdx.x; i < n; i += st) {
        int row = (int)(i / cols);
        d[i] = (row < srows) ? (__bf16)s[i] : (__bf16)0.0f;
    }
}

__global__ void k_init(float* h0, float* enc_outs, long long n_enc, unsigned* bar)
{
    long long i0 = (long long)blockIdx.x * blockDim.x + threadIdx.x;
    long long st = (long long)gridDim.x * blockDim.x;
    for (long long i = i0; i < n_enc; i += st) enc_outs[i] = 0.f;
    for (long long i = i0; i < HS; i += st) h0[i] = 0.f;
    if (i0 < 8) bar[i0] = 0u;
}

// ---------------------------------------------------------------------------
// Persistent seq2seq kernel: 96 WGs x 256 threads (8 wave32 each).
// Encoder: WG t owns rows [16t,16t+16) of each GRU gate block (waves 0..5 do
// the 6 WMMA mat-vec tiles, then 16 threads do the gate math).
// Decoder: 5 device barriers per step (attn logits / context / combine / GRU /
// out logits); softmax+argmax recomputed redundantly per WG to save barriers.
// All mat-vec input vectors are staged once per WG per phase into LDS.
// ---------------------------------------------------------------------------
__global__ __launch_bounds__(TPB) void seq2seq_kernel(Params p)
{
    const int wg   = blockIdx.x;
    const int tid  = threadIdx.x;
    const int wave = tid >> 5;

    __shared__ float s_x[2 * HS];        // staged mat-vec input (concat) vector
    __shared__ float s_gl[8][16];
    __shared__ float s_red[TPB];
    __shared__ int   s_idx[TPB];
    __shared__ float s_attw[136];

    float* ha = p.h0;   // current hidden (read)
    float* hb = p.h1;   // next hidden (write)

    // ------------------------------ encoder ------------------------------
    for (int step = 0; step < p.L; ++step) {
        const int tok = p.tokens[step];
        const float* xe = p.emb_enc + (size_t)tok * HS;
        const int m0 = 16 * wg;
        stage(s_x, xe, HS);
        stage(s_x + HS, ha, HS);
        __syncthreads();
        if (wave < 6) {
            const int blk = (wave < 3) ? wave : wave - 3;
            const __bf16* W = (wave < 3) ? p.w_encWih : p.w_encWhh;
            const float* xv = (wave < 3) ? s_x : s_x + HS;
            wmma_matvec_tile(W, HS, blk * HS + m0, xv, 0, HS, s_gl[wave]);
        }
        __syncthreads();
        if (tid < 16) {
            const int row = m0 + tid;
            float ir = s_gl[0][tid] + p.enc_bih[row];
            float iz = s_gl[1][tid] + p.enc_bih[HS + row];
            float in_= s_gl[2][tid] + p.enc_bih[2 * HS + row];
            float hr = s_gl[3][tid] + p.enc_bhh[row];
            float hz = s_gl[4][tid] + p.enc_bhh[HS + row];
            float hn = s_gl[5][tid] + p.enc_bhh[2 * HS + row];
            float r = 1.f / (1.f + expf(-(ir + hr)));
            float z = 1.f / (1.f + expf(-(iz + hz)));
            float n = tanhf(in_ + r * hn);
            float hv = (1.f - z) * n + z * ha[row];
            hb[row] = hv;
            p.enc_outs[(size_t)step * HS + row] = hv;
        }
        gsync(p.bar);
        { float* t = ha; ha = hb; hb = t; }
    }

    // ------------------------------ decoder ------------------------------
    int cur_tok = 0;  // SOS
    for (int step = 0; step <= p.ML; ++step) {
        if (step > 0) {
            // log-softmax + argmax of previous step's logits (redundant per WG)
            float l = (tid < VS) ? p.logitsO[tid] : -3.4e38f;
            s_red[tid] = l;
            s_idx[tid] = (tid < VS) ? tid : 0x7fffffff;
            __syncthreads();
            for (int s = TPB / 2; s > 0; s >>= 1) {
                if (tid < s) {
                    float bv = s_red[tid + s]; int ib = s_idx[tid + s];
                    float av = s_red[tid];     int ia = s_idx[tid];
                    if (bv > av || (bv == av && ib < ia)) { s_red[tid] = bv; s_idx[tid] = ib; }
                }
                __syncthreads();
            }
            const float mx = s_red[0];
            const int   am = s_idx[0];
            __syncthreads();
            s_red[tid] = (tid < VS) ? expf(l - mx) : 0.f;
            __syncthreads();
            for (int s = TPB / 2; s > 0; s >>= 1) {
                if (tid < s) s_red[tid] += s_red[tid + s];
                __syncthreads();
            }
            const float lse = mx + logf(s_red[0]);
            __syncthreads();
            if (wg == 0 && tid < VS) p.out_logp[(size_t)(step - 1) * VS + tid] = l - lse;
            if (wg == 0 && tid == 0) p.out_tok[step - 1] = (float)am;
            cur_tok = am;
        }
        if (step == p.ML) break;

        const float* xd = p.emb_dec + (size_t)cur_tok * HS;
        const int n_att = p.ML + 1;   // 129

        // ---- D1: attention logits (9 row tiles, K=3072 split over 8 waves) ----
        if (wg < 9) {
            stage(s_x, xd, HS);
            stage(s_x + HS, ha, HS);
            __syncthreads();
            wmma_matvec_tile(p.w_attn, 2 * HS, 16 * wg, s_x, 384 * wave, 384, s_gl[wave]);
            __syncthreads();
            if (tid < 16) {
                const int row = 16 * wg + tid;
                if (row < n_att) {
                    float s = 0.f;
#pragma unroll
                    for (int w = 0; w < 8; ++w) s += s_gl[w][tid];
                    p.logitsA[row] = s + p.attn_b[row];
                }
            }
        }
        gsync(p.bar);

        // ---- D2: softmax (redundant) + 16-column context slice ----
        {
            float v = (tid < n_att) ? p.logitsA[tid] : -3.4e38f;
            s_red[tid] = v;
            __syncthreads();
            for (int s = TPB / 2; s > 0; s >>= 1) {
                if (tid < s) s_red[tid] = fmaxf(s_red[tid], s_red[tid + s]);
                __syncthreads();
            }
            const float mx = s_red[0];
            __syncthreads();
            const float e = (tid < n_att) ? expf(v - mx) : 0.f;
            if (tid < n_att) s_attw[tid] = e;
            s_red[tid] = e;
            __syncthreads();
            for (int s = TPB / 2; s > 0; s >>= 1) {
                if (tid < s) s_red[tid] += s_red[tid + s];
                __syncthreads();
            }
            const float inv = 1.f / s_red[0];
            __syncthreads();
            const int j = 16 * wg + (tid & 15);
            float part = 0.f;
            for (int i = (tid >> 4); i < n_att; i += 16)
                part += s_attw[i] * p.enc_outs[(size_t)i * HS + j];
            s_red[tid] = part;
            __syncthreads();
            for (int s = TPB / 2; s >= 16; s >>= 1) {
                if (tid < s) s_red[tid] += s_red[tid + s];
                __syncthreads();
            }
            if (tid < 16) p.ctx[16 * wg + tid] = s_red[tid] * inv;
        }
        gsync(p.bar);

        // ---- D3: x = relu(comb_W @ [emb;ctx] + b), 1 tile/WG, K split 8 ways ----
        stage(s_x, xd, HS);
        stage(s_x + HS, p.ctx, HS);
        __syncthreads();
        wmma_matvec_tile(p.w_comb, 2 * HS, 16 * wg, s_x, 384 * wave, 384, s_gl[wave]);
        __syncthreads();
        if (tid < 16) {
            const int row = 16 * wg + tid;
            float s = 0.f;
#pragma unroll
            for (int w = 0; w < 8; ++w) s += s_gl[w][tid];
            p.xvec[row] = fmaxf(s + p.comb_b[row], 0.f);
        }
        gsync(p.bar);

        // ---- D4: GRU (6 gate tiles per WG, triple-colocated rows) ----
        {
            const int m0 = 16 * wg;
            stage(s_x, p.xvec, HS);
            stage(s_x + HS, ha, HS);
            __syncthreads();
            if (wave < 6) {
                const int blk = (wave < 3) ? wave : wave - 3;
                const __bf16* W = (wave < 3) ? p.w_decWih : p.w_decWhh;
                const float* xv = (wave < 3) ? s_x : s_x + HS;
                wmma_matvec_tile(W, HS, blk * HS + m0, xv, 0, HS, s_gl[wave]);
            }
            __syncthreads();
            if (tid < 16) {
                const int row = m0 + tid;
                float ir = s_gl[0][tid] + p.dec_bih[row];
                float iz = s_gl[1][tid] + p.dec_bih[HS + row];
                float in_= s_gl[2][tid] + p.dec_bih[2 * HS + row];
                float hr = s_gl[3][tid] + p.dec_bhh[row];
                float hz = s_gl[4][tid] + p.dec_bhh[HS + row];
                float hn = s_gl[5][tid] + p.dec_bhh[2 * HS + row];
                float r = 1.f / (1.f + expf(-(ir + hr)));
                float z = 1.f / (1.f + expf(-(iz + hz)));
                float n = tanhf(in_ + r * hn);
                hb[row] = (1.f - z) * n + z * ha[row];
            }
        }
        gsync(p.bar);
        { float* t = ha; ha = hb; hb = t; }   // ha = new hidden

        // ---- D5: output logits (8 tiles, K=1536 split over 8 waves) ----
        if (wg < 8) {
            stage(s_x, ha, HS);
            __syncthreads();
            wmma_matvec_tile(p.w_out, HS, 16 * wg, s_x, 192 * wave, 192, s_gl[wave]);
            __syncthreads();
            if (tid < 16) {
                const int row = 16 * wg + tid;
                float s = 0.f;
#pragma unroll
                for (int w = 0; w < 8; ++w) s += s_gl[w][tid];
                p.logitsO[row] = s + p.out_b[row];
            }
        }
        gsync(p.bar);
    }
}

// ---------------------------------------------------------------------------
extern "C" void kernel_launch(void* const* d_in, const int* in_sizes, int n_in,
                              void* d_out, int out_size, void* d_ws, size_t ws_size,
                              hipStream_t stream)
{
    (void)n_in; (void)out_size; (void)ws_size;
    const int*   tokens  = (const int*)  d_in[0];
    const float* emb_enc = (const float*)d_in[2];
    const float* enc_Wih = (const float*)d_in[3];
    const float* enc_Whh = (const float*)d_in[4];
    const float* enc_bih = (const float*)d_in[5];
    const float* enc_bhh = (const float*)d_in[6];
    const float* emb_dec = (const float*)d_in[7];
    const float* attn_W  = (const float*)d_in[8];
    const float* attn_b  = (const float*)d_in[9];
    const float* comb_W  = (const float*)d_in[10];
    const float* comb_b  = (const float*)d_in[11];
    const float* dec_Wih = (const float*)d_in[12];
    const float* dec_Whh = (const float*)d_in[13];
    const float* dec_bih = (const float*)d_in[14];
    const float* dec_bhh = (const float*)d_in[15];
    const float* out_W   = (const float*)d_in[16];
    const float* out_b   = (const float*)d_in[17];

    const int L  = in_sizes[0];
    const int ML = in_sizes[8] / (2 * HS) - 1;   // attn_W rows = ML+1

    char* ws = (char*)d_ws;
    size_t off = 0;
    auto take = [&](size_t bytes) -> void* {
        void* r = ws + off;
        off = (off + bytes + 255) & ~(size_t)255;
        return r;
    };
    const size_t szW = (size_t)3 * HS * HS;                 // 3H x H
    __bf16* w_encWih = (__bf16*)take(szW * 2);
    __bf16* w_encWhh = (__bf16*)take(szW * 2);
    __bf16* w_decWih = (__bf16*)take(szW * 2);
    __bf16* w_decWhh = (__bf16*)take(szW * 2);
    __bf16* w_comb   = (__bf16*)take((size_t)HS * 2 * HS * 2);
    __bf16* w_attn   = (__bf16*)take((size_t)144 * 2 * HS * 2);  // padded to 9 tiles
    __bf16* w_out    = (__bf16*)take((size_t)VS * HS * 2);
    float* h0       = (float*)take((size_t)HS * 4);
    float* h1       = (float*)take((size_t)HS * 4);
    float* enc_outs = (float*)take((size_t)(ML + 1) * HS * 4);
    float* logitsA  = (float*)take(160 * 4);
    float* ctx      = (float*)take((size_t)HS * 4);
    float* xvec     = (float*)take((size_t)HS * 4);
    float* logitsO  = (float*)take((size_t)VS * 4);
    unsigned* bar   = (unsigned*)take(256);

    // fp32 -> bf16 weight conversion (L2-resident working set: ~68 MB)
    k_cvt<<<2048, 256, 0, stream>>>(enc_Wih, w_encWih, (long long)szW);
    k_cvt<<<2048, 256, 0, stream>>>(enc_Whh, w_encWhh, (long long)szW);
    k_cvt<<<2048, 256, 0, stream>>>(dec_Wih, w_decWih, (long long)szW);
    k_cvt<<<2048, 256, 0, stream>>>(dec_Whh, w_decWhh, (long long)szW);
    k_cvt<<<2048, 256, 0, stream>>>(comb_W, w_comb, (long long)HS * 2 * HS);
    k_cvt<<<128,  256, 0, stream>>>(out_W,  w_out,  (long long)VS * HS);
    k_cvt_pad<<<512, 256, 0, stream>>>(attn_W, w_attn, ML + 1, 144, 2 * HS);
    k_init<<<256, 256, 0, stream>>>(h0, enc_outs, (long long)(ML + 1) * HS, bar);

    Params p;
    p.tokens = tokens; p.L = L; p.ML = ML;
    p.emb_enc = emb_enc; p.enc_bih = enc_bih; p.enc_bhh = enc_bhh;
    p.emb_dec = emb_dec; p.attn_b = attn_b; p.comb_b = comb_b;
    p.dec_bih = dec_bih; p.dec_bhh = dec_bhh; p.out_b = out_b;
    p.w_encWih = w_encWih; p.w_encWhh = w_encWhh;
    p.w_decWih = w_decWih; p.w_decWhh = w_decWhh;
    p.w_comb = w_comb; p.w_attn = w_attn; p.w_out = w_out;
    p.h0 = h0; p.h1 = h1; p.enc_outs = enc_outs;
    p.logitsA = logitsA; p.ctx = ctx; p.xvec = xvec; p.logitsO = logitsO;
    p.out_logp = (float*)d_out;
    p.out_tok  = (float*)d_out + (size_t)ML * VS;
    p.bar = bar;

    seq2seq_kernel<<<NWG, TPB, 0, stream>>>(p);
}
